// CNN_LSTM_70308614635607
// MI455X (gfx1250) — compile-verified, matrix operand order checked by reading, not run on hardware
//
#include <hip/hip_runtime.h>

typedef unsigned short u16t;
typedef unsigned int   u32t;
typedef __attribute__((ext_vector_type(8)))  unsigned short u16x8;
typedef __attribute__((ext_vector_type(16))) __bf16         bf16x16;
typedef __attribute__((ext_vector_type(8)))  float          f32x8;

#define NFRAMES 4096
#define NPOS    121
#define CH      64
#define FRAME_E (NPOS*CH)   /* 7744 bf16 elements per frame */
#define FPW     8           /* frames per workgroup in conv */

// ---- CDNA5 feature detection -------------------------------------------------
#if defined(__has_builtin)
#  if __has_builtin(__builtin_amdgcn_tensor_load_to_lds) && __has_builtin(__builtin_amdgcn_s_wait_tensorcnt)
#    define HAVE_TDM 1
#  elif __has_builtin(__builtin_amdgcn_global_load_async_to_lds_b128)
#    define HAVE_ASYNC128 1
#  endif
#  if __has_builtin(__builtin_amdgcn_global_load_async_to_lds_b32)
#    define HAVE_ASYNC32 1
#  endif
#endif

#if __has_include(<hip/amd_detail/amd_gfx1250_TDM.h>)
#  define TDM_SIX_ARG 1   /* therock headers -> 6-arg tensor builtin */
#endif

#if defined(__has_builtin)
#  if __has_builtin(__builtin_amdgcn_s_wait_asynccnt)
#    define WAIT_ASYNC(n) __builtin_amdgcn_s_wait_asynccnt(n)
#  else
#    define WAIT_ASYNC(n) asm volatile("s_wait_asynccnt %0" :: "i"(n) : "memory")
#  endif
#endif

// async-LDS builtins take int-typed AS pointers (per hipcc diagnostic)
typedef __attribute__((address_space(1))) int glob_int_t;
typedef __attribute__((address_space(3))) int lds_int_t;

#if defined(HAVE_TDM)
typedef __attribute__((ext_vector_type(4))) unsigned int uint32x4_t_;
typedef __attribute__((ext_vector_type(8))) int          int32x8_t_;
typedef __attribute__((ext_vector_type(4))) int          int32x4_t_;
// 1-D contiguous DMA (global -> LDS), bytes multiple of 8, <= 64K 8B-units.
__device__ __forceinline__ void tdm_copy_1d(const void* gsrc, void* ldst, unsigned bytes) {
  unsigned units = bytes >> 3;                       // data_size = 8B units
  unsigned long long ga = (unsigned long long)gsrc;
  uint32x4_t_ g0;
  g0[0] = 1u;                                        // count=1, user mode
  g0[1] = (unsigned)(unsigned long long)ldst;        // lds byte offset (low 32 of flat)
  g0[2] = (unsigned)ga;                              // global_addr[31:0]
  g0[3] = ((unsigned)((ga >> 32) & 0x01FFFFFFu)) | (2u << 30);  // addr[56:32] | type=2
  int32x8_t_ g1;
  g1[0] = 0x00030000;                                // wg_mask=0, data_size=3 (8B)
  g1[1] = (int)((units & 0xFFFFu) << 16);            // tensor_dim0[15:0]
  g1[2] = (int)((units >> 16) | (1u << 16));         // tensor_dim0[31:16] | tensor_dim1=1
  g1[3] = (int)(units << 16);                        // tile_dim0
  g1[4] = 1;                                         // tile_dim1=1, tile_dim2=0
  g1[5] = (int)units;                                // tensor_dim0_stride[31:0]
  g1[6] = 0;
  g1[7] = 0;
  int32x4_t_ gz; gz[0]=0; gz[1]=0; gz[2]=0; gz[3]=0;
#if defined(TDM_SIX_ARG)
  int32x8_t_ gz8; gz8[0]=0; gz8[1]=0; gz8[2]=0; gz8[3]=0; gz8[4]=0; gz8[5]=0; gz8[6]=0; gz8[7]=0;
  __builtin_amdgcn_tensor_load_to_lds(g0, g1, gz, gz, gz8, 0);
#else
  __builtin_amdgcn_tensor_load_to_lds(g0, g1, gz, gz, 0);
#endif
}
#endif // HAVE_TDM

#if defined(HAVE_ASYNC128)
// per-lane 16B async copy of one frame (968 chunks), uniform 4 instructions/wave
__device__ __forceinline__ void async_frame(const u16t* gsrc, u16t* ldst, int tid) {
  const char* g = (const char*)gsrc;
  char* l = (char*)ldst;
#pragma unroll
  for (int it = 0; it < 4; ++it) {
    int e = tid + it * 256;
    e = e < 968 ? e : 967;                         // clamp -> full EXEC, uniform count
    __builtin_amdgcn_global_load_async_to_lds_b128(
        (glob_int_t*)(unsigned long long)(g + (size_t)e * 16),
        (lds_int_t*)(unsigned)(unsigned long long)(l + e * 16), 0, 0);
  }
}
#endif

__device__ __forceinline__ u16t f2bf(float f) {
  u32t u = __builtin_bit_cast(u32t, f);
  u32t r = u + 0x7FFFu + ((u >> 16) & 1u);   // round-to-nearest-even
  return (u16t)(r >> 16);
}
__device__ __forceinline__ float bf2f(u16t h) {
  u32t u = ((u32t)h) << 16;
  return __builtin_bit_cast(float, u);
}
union FragU { u16x8 h[2]; bf16x16 v; };
__device__ __forceinline__ bf16x16 mkfrag(u16x8 lo, u16x8 hi) {
  FragU u; u.h[0] = lo; u.h[1] = hi; return u.v;
}
__device__ __forceinline__ float sigm(float x) { return 1.f / (1.f + __expf(-x)); }

// ---------------- packing kernels (fp32 -> bf16, layout changes) ----------------

// x [4096][9][121] f32 -> act [4096][121][64] bf16 (channels 9..63 = 0)
__global__ void pack_input(const float* __restrict__ x, u16t* __restrict__ act) {
  int idx = blockIdx.x * 256 + threadIdx.x;            // one uint (2 channels) each
  if (idx >= NFRAMES * NPOS * 32) return;
  int frame = idx / (NPOS * 32);
  int r     = idx % (NPOS * 32);
  int pos = r >> 5, cp = r & 31, c0 = cp * 2;
  const float* xb = x + (size_t)frame * 9 * NPOS;
  float f0 = (c0     < 9) ? xb[(size_t)c0 * NPOS + pos]       : 0.f;
  float f1 = (c0 + 1 < 9) ? xb[(size_t)(c0 + 1) * NPOS + pos] : 0.f;
  ((u32t*)act)[idx] = ((u32t)f2bf(f1) << 16) | (u32t)f2bf(f0);
}

// conv0_w [64][9][3][3], conv_ws [7][64][64][3][3] -> wpack [8][tap=9][o=64][i=64] bf16
__global__ void pack_convw(const float* __restrict__ w0, const float* __restrict__ wrest,
                           u16t* __restrict__ wpack) {
  int idx = blockIdx.x * 256 + threadIdx.x;
  if (idx >= 8 * 9 * 64 * 64) return;
  int L  = idx / 36864;
  int r  = idx % 36864;
  int tap = r / 4096;
  int rr  = r % 4096;
  int o = rr >> 6, i = rr & 63;
  float v;
  if (L == 0) v = (i < 9) ? w0[(size_t)o * 81 + i * 9 + tap] : 0.f;
  else        v = wrest[(size_t)(L - 1) * 36864 + (size_t)o * 576 + i * 9 + tap];
  wpack[idx] = f2bf(v);
}

__global__ void pack_bias(const float* __restrict__ c0b, const float* __restrict__ cbs,
                          float* __restrict__ biasAll) {
  int idx = blockIdx.x * 256 + threadIdx.x;
  if (idx >= 512) return;
  int L = idx >> 6, c = idx & 63;
  biasAll[idx] = (L == 0) ? c0b[c] : cbs[(L - 1) * 64 + c];
}

// fc_w [64][c*121+pos] -> fcwt [64][pos*64+c] bf16
__global__ void pack_fcw(const float* __restrict__ fcw, u16t* __restrict__ fcwt) {
  int idx = blockIdx.x * 256 + threadIdx.x;
  if (idx >= 64 * FRAME_E) return;
  int u  = idx / FRAME_E;
  int k2 = idx % FRAME_E;
  int pos = k2 >> 6, c = k2 & 63;
  fcwt[idx] = f2bf(fcw[(size_t)u * FRAME_E + (size_t)c * NPOS + pos]);
}

// wlstm [n=256][k=128] bf16 : k<64 -> w_ih[n][k] ; k>=64 -> w_hh[n][k-64]
__global__ void pack_wlstm(const float* __restrict__ wih, const float* __restrict__ whh,
                           u16t* __restrict__ wlstm) {
  int idx = blockIdx.x * 256 + threadIdx.x;
  if (idx >= 256 * 128) return;
  int n = idx >> 7, k = idx & 127;
  wlstm[idx] = f2bf(k < 64 ? wih[(size_t)n * 64 + k] : whh[(size_t)n * 64 + (k - 64)]);
}

// ---------------- conv layer: implicit GEMM via WMMA bf16 + TDM pipeline ----------------
// one workgroup (8 waves) per 8 frames; per frame wave w owns M-tile w (16 spatial pos)
__global__ __launch_bounds__(256) void conv_wmma(const u16t* __restrict__ actIn,
                                                 u16t* __restrict__ actOut,
                                                 const u16t* __restrict__ wL,
                                                 const float* __restrict__ biasL) {
  __shared__ __align__(16) u16t  sIn[2][FRAME_E];   // double-buffered [pos][c] bf16
  __shared__ __align__(16) u16t  sW[9 * 64 * 64];   // [tap][cout][cin] bf16
  __shared__ float sBias[64];
  __shared__ __align__(16) u16t  sZero[64];
  int tid = threadIdx.x;
  int f0  = blockIdx.x * FPW;

  if (tid < 32) ((u32t*)sZero)[tid] = 0u;
  if (tid < 64) sBias[tid] = biasL[tid];
  if (f0 + FPW < NFRAMES)
    __builtin_prefetch(actIn + (size_t)(f0 + FPW) * FRAME_E, 0, 1);

#if defined(HAVE_TDM)
  if (tid < 32) {                                   // wave 0 drives the TDM
    tdm_copy_1d(wL, sW, 9 * 64 * 64 * 2);
    tdm_copy_1d(actIn + (size_t)f0 * FRAME_E, &sIn[0][0], FRAME_E * 2);
  }
#elif defined(HAVE_ASYNC128)
  {
    const char* g = (const char*)wL;
    char* l = (char*)sW;
#pragma unroll
    for (int it = 0; it < 18; ++it) {               // 73728/16 = 4608 chunks, uniform
      int e = tid + it * 256;
      __builtin_amdgcn_global_load_async_to_lds_b128(
          (glob_int_t*)(unsigned long long)(g + (size_t)e * 16),
          (lds_int_t*)(unsigned)(unsigned long long)(l + e * 16), 0, 0);
    }
    async_frame(actIn + (size_t)f0 * FRAME_E, &sIn[0][0], tid);
  }
#else
  {
    const u32t* gw = (const u32t*)wL;
    u32t* sW32 = (u32t*)sW;
    for (int e = tid; e < 18432; e += 256) sW32[e] = gw[e];
  }
#endif

  int lane = tid & 31, wv = tid >> 5;
  int half = lane >> 4, lm = lane & 15;      // A: row M = lm; half selects K sub-lanes
  int mtile = wv;                            // 0..7 (121 rows -> 8 tiles, last partial)
  int p = mtile * 16 + lm;
  bool valid = (p < NPOS);
  int y = valid ? (p / 11) : 0;
  int x = valid ? (p % 11) : 0;

  for (int f = 0; f < FPW; ++f) {
    const u16t* sInF = &sIn[f & 1][0];

#if defined(HAVE_TDM)
    if (f + 1 < FPW) {
      if (tid < 32) {
        tdm_copy_1d(actIn + (size_t)(f0 + f + 1) * FRAME_E, &sIn[(f + 1) & 1][0], FRAME_E * 2);
        __builtin_amdgcn_s_wait_tensorcnt(1);   // frame f landed (in-order), f+1 in flight
      }
    } else {
      if (tid < 32) __builtin_amdgcn_s_wait_tensorcnt(0);
    }
#elif defined(HAVE_ASYNC128)
    if (f + 1 < FPW) {
      async_frame(actIn + (size_t)(f0 + f + 1) * FRAME_E, &sIn[(f + 1) & 1][0], tid);
      WAIT_ASYNC(4);                            // only f+1's 4 loads still outstanding
    } else {
      WAIT_ASYNC(0);
    }
#else
    {
      const u32t* gin = (const u32t*)(actIn + (size_t)(f0 + f) * FRAME_E);
      u32t* d = (u32t*)sInF;
      for (int e = tid; e < FRAME_E / 2; e += 256) d[e] = gin[e];
    }
#endif
    __syncthreads();                            // data visible to all waves

    f32x8 acc[4];
#pragma unroll
    for (int q = 0; q < 4; ++q)
#pragma unroll
      for (int j = 0; j < 8; ++j) acc[q][j] = 0.f;

    for (int tap = 0; tap < 9; ++tap) {
      int iy = y + tap / 3 - 1;
      int ix = x + tap % 3 - 1;
      bool inb = valid && iy >= 0 && iy < 11 && ix >= 0 && ix < 11;
      const u16t* asrc = inb ? (sInF + (iy * 11 + ix) * 64) : sZero;
#pragma unroll
      for (int c0 = 0; c0 < 64; c0 += 32) {
        u16x8 alo = *(const u16x8*)(asrc + c0 + half * 8);
        u16x8 ahi = *(const u16x8*)(asrc + c0 + 16 + half * 8);
        bf16x16 af = mkfrag(alo, ahi);
#pragma unroll
        for (int nt = 0; nt < 4; ++nt) {
          const u16t* bsrc = sW + ((tap * 64 + nt * 16 + lm) * 64) + c0 + half * 8;
          bf16x16 bf = mkfrag(*(const u16x8*)bsrc, *(const u16x8*)(bsrc + 16));
          acc[nt] = __builtin_amdgcn_wmma_f32_16x16x32_bf16(
              false, af, false, bf, (short)0, acc[nt], false, false);
        }
      }
    }

    u16t* gout = actOut + (size_t)(f0 + f) * FRAME_E;
#pragma unroll
    for (int nt = 0; nt < 4; ++nt) {
      int o = nt * 16 + lm;                    // D: N = lane&15
      float bb = sBias[o];
#pragma unroll
      for (int j = 0; j < 8; ++j) {            // D: M = j + 8*half
        int pp = mtile * 16 + half * 8 + j;
        if (pp < NPOS) {
          float vv = acc[nt][j] + bb;
          gout[pp * 64 + o] = f2bf(vv > 0.f ? vv : 0.f);
        }
      }
    }
    __syncthreads();                           // buffer free for next prefetch
  }
}

// ---------------- FC: feats = relu(act @ fcwt^T + b), M=4096 N=64 K=7744 ----------------
__global__ __launch_bounds__(256) void fc_wmma(const u16t* __restrict__ act,
                                               const u16t* __restrict__ fcwt,
                                               const float* __restrict__ fcb,
                                               u16t* __restrict__ featsbf) {
  int tid = threadIdx.x, lane = tid & 31, wv = tid >> 5;
  int half = lane >> 4, lm = lane & 15;
  int mtile = blockIdx.x * 2 + (wv >> 2);      // 0..255
  int ntile = wv & 3;                          // 0..3
  const u16t* arow = act  + (size_t)(mtile * 16 + lm) * FRAME_E;
  const u16t* brow = fcwt + (size_t)(ntile * 16 + lm) * FRAME_E;
  f32x8 acc;
#pragma unroll
  for (int j = 0; j < 8; ++j) acc[j] = 0.f;
  for (int c0 = 0; c0 < FRAME_E; c0 += 32) {   // 242 WMMAs
    u16x8 alo = *(const u16x8*)(arow + c0 + half * 8);
    u16x8 ahi = *(const u16x8*)(arow + c0 + 16 + half * 8);
    u16x8 blo = *(const u16x8*)(brow + c0 + half * 8);
    u16x8 bhi = *(const u16x8*)(brow + c0 + 16 + half * 8);
    acc = __builtin_amdgcn_wmma_f32_16x16x32_bf16(
        false, mkfrag(alo, ahi), false, mkfrag(blo, bhi), (short)0, acc, false, false);
  }
  int u = ntile * 16 + lm;
  float bb = fcb[u];
#pragma unroll
  for (int j = 0; j < 8; ++j) {
    int fr = mtile * 16 + half * 8 + j;
    float vv = acc[j] + bb;
    featsbf[(size_t)fr * 64 + u] = f2bf(vv > 0.f ? vv : 0.f);
  }
}

// ---------------- LSTM: fused-K gate GEMM [32x128]@[128x256] per step, WMMA ----------------
__global__ __launch_bounds__(256) void lstm_wmma(const u16t* __restrict__ featsbf,
                                                 const u16t* __restrict__ wlstm,
                                                 const float* __restrict__ bih,
                                                 const float* __restrict__ bhh,
                                                 float* __restrict__ lstmo) {
  __shared__ __align__(16) u16t  sA[32 * 128];   // [b][ x(64) | h(64) ] bf16
  __shared__ float sG[32 * 256];                 // gates i,f,g,o
  __shared__ float sH[32 * 64];
  __shared__ float sC[32 * 64];
  __shared__ float sBias[256];
  int tid = threadIdx.x;
  sBias[tid] = bih[tid] + bhh[tid];
  for (int i = tid; i < 2048; i += 256) { sH[i] = 0.f; sC[i] = 0.f; }
  __syncthreads();
  int lane = tid & 31, wv = tid >> 5, half = lane >> 4, lm = lane & 15;
  int mtile = wv >> 2;               // 0..1
  int ntb   = (wv & 3) * 4;          // 4 N-tiles per wave
  for (int t = 0; t < 128; ++t) {
    // stage x_t (async LDS path when available) and h_{t-1} into sA
#if defined(HAVE_ASYNC32)
#pragma unroll
    for (int it = 0; it < 4; ++it) {             // 1024 u32 chunks, uniform count
      int e = tid + it * 256;
      int b = e >> 5, kp = e & 31;
      const char* g = (const char*)(featsbf + ((size_t)(b * 128 + t)) * 64 + kp * 2);
      char* l = (char*)(sA + b * 128 + kp * 2);
      __builtin_amdgcn_global_load_async_to_lds_b32(
          (glob_int_t*)(unsigned long long)g,
          (lds_int_t*)(unsigned)(unsigned long long)l, 0, 0);
    }
    for (int i = tid; i < 2048; i += 256) {
      int b = i >> 6, k = i & 63;
      sA[b * 128 + 64 + k] = f2bf(sH[b * 64 + k]);
    }
    WAIT_ASYNC(0);
#else
    for (int i = tid; i < 2048; i += 256) {
      int b = i >> 6, k = i & 63;
      sA[b * 128 + k]      = featsbf[((size_t)(b * 128 + t)) * 64 + k];
      sA[b * 128 + 64 + k] = f2bf(sH[b * 64 + k]);
    }
#endif
    __syncthreads();
    f32x8 acc[4];
#pragma unroll
    for (int q = 0; q < 4; ++q)
#pragma unroll
      for (int j = 0; j < 8; ++j) acc[q][j] = 0.f;
    const u16t* arow = sA + (mtile * 16 + lm) * 128;
#pragma unroll
    for (int c0 = 0; c0 < 128; c0 += 32) {
      bf16x16 af = mkfrag(*(const u16x8*)(arow + c0 + half * 8),
                          *(const u16x8*)(arow + c0 + 16 + half * 8));
#pragma unroll
      for (int q = 0; q < 4; ++q) {
        const u16t* brow = wlstm + (size_t)((ntb + q) * 16 + lm) * 128 + c0 + half * 8;
        acc[q] = __builtin_amdgcn_wmma_f32_16x16x32_bf16(
            false, af, false, mkfrag(*(const u16x8*)brow, *(const u16x8*)(brow + 16)),
            (short)0, acc[q], false, false);
      }
    }
#pragma unroll
    for (int q = 0; q < 4; ++q) {
      int n = (ntb + q) * 16 + lm;
      float bb = sBias[n];
#pragma unroll
      for (int j = 0; j < 8; ++j) {
        int b = mtile * 16 + half * 8 + j;
        sG[b * 256 + n] = acc[q][j] + bb;
      }
    }
    __syncthreads();
    for (int i = tid; i < 2048; i += 256) {
      int b = i >> 6, u = i & 63;
      float ig = sG[b * 256 + u],        fg = sG[b * 256 + 64 + u];
      float gg = sG[b * 256 + 128 + u],  og = sG[b * 256 + 192 + u];
      float c = sigm(fg) * sC[i] + sigm(ig) * tanhf(gg);
      float h = sigm(og) * tanhf(c);
      sC[i] = c; sH[i] = h;
      lstmo[((size_t)(b * 128 + t)) * 64 + u] = h;
    }
    __syncthreads();
  }
}

// ---------------- heads: p = relu(h@p1^T+b)@p2^T+b2 ; v likewise (tiny, VALU) ----------------
__global__ __launch_bounds__(256) void heads_kernel(const float* __restrict__ lstmo,
    const float* __restrict__ p1w, const float* __restrict__ p1b,
    const float* __restrict__ p2w, const float* __restrict__ p2b,
    const float* __restrict__ v1w, const float* __restrict__ v1b,
    const float* __restrict__ v2w, const float* __restrict__ v2b,
    float* __restrict__ out) {
  __shared__ u16t  sP1[128 * 64];
  __shared__ u16t  sV1[128 * 64];
  __shared__ float sP2[6 * 128];
  __shared__ float sV2[128];
  __shared__ float sP1b[128];
  __shared__ float sV1b[128];
  __shared__ float sP2b[8];
  __shared__ float sV2b;
  int tid = threadIdx.x;
  for (int i = tid; i < 8192; i += 256) { sP1[i] = f2bf(p1w[i]); sV1[i] = f2bf(v1w[i]); }
  for (int i = tid; i < 768; i += 256) sP2[i] = p2w[i];
  if (tid < 128) { sP1b[tid] = p1b[tid]; sV1b[tid] = v1b[tid]; sV2[tid] = v2w[tid]; }
  if (tid < 6)  sP2b[tid] = p2b[tid];
  if (tid == 0) sV2b = v2b[0];
  __syncthreads();
  int r = blockIdx.x * 256 + tid;          // 0..4095 = b*128+t
  float h[64];
#pragma unroll
  for (int k = 0; k < 64; ++k) h[k] = lstmo[(size_t)r * 64 + k];
  float pa[6];
#pragma unroll
  for (int a = 0; a < 6; ++a) pa[a] = sP2b[a];
  float va = sV2b;
  for (int u2 = 0; u2 < 128; ++u2) {
    float s = sP1b[u2], sv = sV1b[u2];
    const u16t* wp  = sP1 + u2 * 64;
    const u16t* wvv = sV1 + u2 * 64;
#pragma unroll 8
    for (int k = 0; k < 64; ++k) { s += h[k] * bf2f(wp[k]); sv += h[k] * bf2f(wvv[k]); }
    s  = s  > 0.f ? s  : 0.f;
    sv = sv > 0.f ? sv : 0.f;
#pragma unroll
    for (int a = 0; a < 6; ++a) pa[a] += s * sP2[a * 128 + u2];
    va += sv * sV2[u2];
  }
#pragma unroll
  for (int a = 0; a < 6; ++a) out[(size_t)r * 6 + a] = pa[a];
  out[24576 + r] = va;
}

// ---------------- launch ----------------
extern "C" void kernel_launch(void* const* d_in, const int* in_sizes, int n_in,
                              void* d_out, int out_size, void* d_ws, size_t ws_size,
                              hipStream_t stream) {
  const float* x       = (const float*)d_in[0];
  const float* conv0_w = (const float*)d_in[1];
  const float* conv0_b = (const float*)d_in[2];
  const float* conv_ws = (const float*)d_in[3];
  const float* conv_bs = (const float*)d_in[4];
  const float* fc_w    = (const float*)d_in[5];
  const float* fc_b    = (const float*)d_in[6];
  const float* w_ih    = (const float*)d_in[7];
  const float* w_hh    = (const float*)d_in[8];
  const float* b_ih    = (const float*)d_in[9];
  const float* b_hh    = (const float*)d_in[10];
  const float* p1_w    = (const float*)d_in[11];
  const float* p1_b    = (const float*)d_in[12];
  const float* p2_w    = (const float*)d_in[13];
  const float* p2_b    = (const float*)d_in[14];
  const float* v1_w    = (const float*)d_in[15];
  const float* v1_b    = (const float*)d_in[16];
  const float* v2_w    = (const float*)d_in[17];
  const float* v2_b    = (const float*)d_in[18];

  char* ws = (char*)d_ws;
  u16t*  actA    = (u16t*) (ws + 0);                 // 63,438,848 B
  u16t*  actB    = (u16t*) (ws + 63438848);          // 63,438,848 B
  u16t*  wpack   = (u16t*) (ws + 126877696);         //    589,824 B
  float* biasAll = (float*)(ws + 127467520);         //      2,048 B
  u16t*  fcwt    = (u16t*) (ws + 127469568);         //    991,232 B
  u16t*  wlstm   = (u16t*) (ws + 128460800);         //     65,536 B
  u16t*  featsbf = (u16t*) (ws + 128526336);         //    524,288 B
  float* lstmo   = (float*)(ws + 129050624);         //  1,048,576 B

  pack_input <<<61952, 256, 0, stream>>>(x, actA);
  pack_convw <<<1152,  256, 0, stream>>>(conv0_w, conv_ws, wpack);
  pack_bias  <<<2,     256, 0, stream>>>(conv0_b, conv_bs, biasAll);
  pack_fcw   <<<1936,  256, 0, stream>>>(fc_w, fcwt);
  pack_wlstm <<<128,   256, 0, stream>>>(w_ih, w_hh, wlstm);

  const u16t* cin = actA;
  u16t* cout_ = actB;
  for (int L = 0; L < 8; ++L) {
    conv_wmma<<<NFRAMES / FPW, 256, 0, stream>>>(cin, cout_, wpack + (size_t)L * 36864,
                                                 biasAll + L * 64);
    u16t* tmp = (u16t*)cin; cin = cout_; cout_ = tmp;
  }

  fc_wmma   <<<128, 256, 0, stream>>>(cin, fcwt, fc_b, featsbf);
  lstm_wmma <<<1,   256, 0, stream>>>(featsbf, wlstm, b_ih, b_hh, lstmo);
  heads_kernel<<<16, 256, 0, stream>>>(lstmo, p1_w, p1_b, p2_w, p2_b,
                                       v1_w, v1_b, v2_w, v2_b, (float*)d_out);
}
